// MultiHeadAttention_51694226375516
// MI455X (gfx1250) — matrix operand order repaired and checked by reading, not compiled, not run on hardware
//
#include <hip/hip_runtime.h>

// ---------------------------------------------------------------------------
// MI455X (gfx1250) causal multi-head attention forward.
// wave32, WMMA bf16 16x16x32 with f32 accumulation for every matmul.
// GEMM: 32x64 output per wave (B fragments reused across two M tiles).
// Attention: 32 queries per wave (K/V fragments reused across two Q tiles).
// ---------------------------------------------------------------------------

typedef __bf16 bf16;
typedef __attribute__((ext_vector_type(4)))  __bf16 bf16x4;
typedef __attribute__((ext_vector_type(8)))  __bf16 bf16x8;
typedef __attribute__((ext_vector_type(16))) __bf16 bf16x16;
typedef __attribute__((ext_vector_type(8)))  float  f32x8;

#define KB 4
#define KS 2048
#define KD 1024
#define KH 16
#define KDK 64
#define KM (KB * KS)          // 8192 token rows
#define ATT_SCALE 0.125f      // 1/sqrt(64)

static __device__ __forceinline__ bf16x16 join8(bf16x8 a, bf16x8 b) {
  return __builtin_shufflevector(a, b, 0,1,2,3,4,5,6,7,8,9,10,11,12,13,14,15);
}
static __device__ __forceinline__ f32x8 wmma_bf16(bf16x16 a, bf16x16 b, f32x8 c) {
  // (neg_a, A, neg_b, B, c_mod, C, reuse_a, reuse_b)
  return __builtin_amdgcn_wmma_f32_16x16x32_bf16(false, a, false, b, (short)0, c,
                                                 false, false);
}

// ---------------------------------------------------------------------------
// fp32 -> bf16 conversion, 4 elements per thread (b64 stores).
// ---------------------------------------------------------------------------
__global__ void cvt_f32_bf16_x4(const float4* __restrict__ src,
                                bf16x4* __restrict__ dst, int n4) {
  int i = blockIdx.x * blockDim.x + threadIdx.x;
  if (i >= n4) return;
  float4 v = src[i];
  bf16x4 r = { (bf16)v.x, (bf16)v.y, (bf16)v.z, (bf16)v.w };
  dst[i] = r;
}

// ---------------------------------------------------------------------------
// Y[M,N] = Xbf[M,K] @ Wbf[N,K]^T + bias   (K = N = 1024, M = 8192, fp32 out)
// One wave -> 32x64 output tile: 2 M-tiles x 4 N-tiles, 8 WMMA per 32-k step.
// A fragment (16-bit A 16x32): lane = {row=lane&15, half=lane>>4},
//   elems 0..7 = K kk+half*8+0..7, elems 8..15 = K kk+16+half*8+0..7.
// B fragment (16-bit B 32x16): lane n = lane&15, same K order as A.
// Each B fragment feeds both M tiles (halves B-operand traffic per FLOP).
// ---------------------------------------------------------------------------
__global__ __launch_bounds__(256)
void gemm_bf16_nt(const bf16* __restrict__ X, const bf16* __restrict__ W,
                  const float* __restrict__ bias, float* __restrict__ Y) {
  const int lane = threadIdx.x & 31;
  const int wave = threadIdx.x >> 5;
  const int row  = lane & 15;
  const int half = lane >> 4;
  const int m0   = (blockIdx.x * 8 + wave) * 32;
  const int n0   = blockIdx.y * 64;

  f32x8 acc[2][4];
#pragma unroll
  for (int nt = 0; nt < 4; ++nt) {
    float bcol = bias[n0 + nt * 16 + row];
#pragma unroll
    for (int u = 0; u < 2; ++u)
#pragma unroll
      for (int i = 0; i < 8; ++i) acc[u][nt][i] = bcol;
  }

  const bf16* xrow0 = X + (size_t)(m0 + row) * KD;
  const bf16* xrow1 = X + (size_t)(m0 + 16 + row) * KD;
  for (int kk = 0; kk < KD; kk += 32) {
    bf16x16 a0 = join8(*(const bf16x8*)(xrow0 + kk + half * 8),
                       *(const bf16x8*)(xrow0 + kk + 16 + half * 8));
    bf16x16 a1 = join8(*(const bf16x8*)(xrow1 + kk + half * 8),
                       *(const bf16x8*)(xrow1 + kk + 16 + half * 8));
#pragma unroll
    for (int nt = 0; nt < 4; ++nt) {
      const bf16* wrow = W + (size_t)(n0 + nt * 16 + row) * KD + kk + half * 8;
      bf16x16 bfrag = join8(*(const bf16x8*)wrow, *(const bf16x8*)(wrow + 16));
      acc[0][nt] = wmma_bf16(a0, bfrag, acc[0][nt]);
      acc[1][nt] = wmma_bf16(a1, bfrag, acc[1][nt]);
    }
  }

  // C layout: slot i holds row m0 + u*16 + i + 8*half, col n0 + nt*16 + (lane&15)
#pragma unroll
  for (int u = 0; u < 2; ++u)
#pragma unroll
    for (int nt = 0; nt < 4; ++nt)
#pragma unroll
      for (int i = 0; i < 8; ++i)
        Y[(size_t)(m0 + u * 16 + i + 8 * half) * KD + n0 + nt * 16 + row] =
            acc[u][nt][i];
}

// ---------------------------------------------------------------------------
// RoPE (faithful to reference rotation) + pack to bf16.
// mode=1: out[b][h][s][d] = x*cos + rot(x)*sin   (Q, K)
// mode=0: out[b][h][d][s] = x                    (V, transposed for PV B-frags)
// ---------------------------------------------------------------------------
__global__ void rope_pack(const float* __restrict__ Xf, bf16* __restrict__ Out,
                          int mode) {
  size_t idx = (size_t)blockIdx.x * blockDim.x + threadIdx.x;  // exactly M*D
  int d = (int)(idx & 63);
  int h = (int)((idx >> 6) & 15);
  int s = (int)((idx >> 10) & 2047);
  int b = (int)(idx >> 21);
  float v = Xf[idx];
  if (mode) {
    size_t base = idx - (size_t)d;  // start of this (b,s,h) 64-wide head row
    // x_rot = cat([-x[1::2], x[::2]]):  d<32 -> -x[2d+1], d>=32 -> x[2d-64]
    float part = (d < 32) ? -Xf[base + 2 * d + 1] : Xf[base + 2 * d - 64];
    int fi = d & 31;
    float inv = __expf(-(float)fi * (9.210340371976184f / 32.0f));  // 10000^(-fi/32)
    float sn, cs;
    __sincosf((float)s * inv, &sn, &cs);
    float o = v * cs + part * sn;
    Out[(((size_t)b * KH + h) * KS + s) * KDK + d] = (bf16)o;
  } else {
    Out[(((size_t)b * KH + h) * KDK + d) * KS + s] = (bf16)v;
  }
}

// ---------------------------------------------------------------------------
// Flash attention: one wave = two 16-query tiles (32 queries) of one (b,h).
// Q,K in [B,H,S,64] bf16 ; V transposed [B,H,64,S] bf16 ; ctx -> [B,S,D] bf16.
// K and V fragments are loaded once per 32-key step and reused by both query
// tiles. Causal mask applied analytically (reference mask is a constant tril).
// ---------------------------------------------------------------------------
__global__ __launch_bounds__(256)
void flash_attn(const bf16* __restrict__ Qb, const bf16* __restrict__ Kb,
                const bf16* __restrict__ Vt, bf16* __restrict__ Ctx) {
  __shared__ bf16 ldsP[8][2 * 16 * 32];  // wave-private P tiles (C->A relayout)

  const int lane = threadIdx.x & 31;
  const int wave = threadIdx.x >> 5;
  const int row  = lane & 15;
  const int half = lane >> 4;
  const int bh   = blockIdx.y;
  const int b    = bh >> 4, h = bh & 15;
  const int q0   = (blockIdx.x * 8 + wave) * 32;

  const bf16* Qp = Qb + (size_t)bh * KS * KDK;
  const bf16* Kp = Kb + (size_t)bh * KS * KDK;
  const bf16* Vp = Vt + (size_t)bh * KDK * KS;

  // Q fragments: two query tiles x two k-steps over head dims
  bf16x16 aq[2][2];
#pragma unroll
  for (int u = 0; u < 2; ++u) {
    const bf16* qrow = Qp + (size_t)(q0 + u * 16 + row) * KDK;
    aq[u][0] = join8(*(const bf16x8*)(qrow + half * 8),
                     *(const bf16x8*)(qrow + 16 + half * 8));
    aq[u][1] = join8(*(const bf16x8*)(qrow + 32 + half * 8),
                     *(const bf16x8*)(qrow + 48 + half * 8));
  }

  f32x8 acc[2][4];
  float mrow[2][8], lrow[2][8];
#pragma unroll
  for (int u = 0; u < 2; ++u) {
#pragma unroll
    for (int nt = 0; nt < 4; ++nt)
#pragma unroll
      for (int i = 0; i < 8; ++i) acc[u][nt][i] = 0.0f;
#pragma unroll
    for (int i = 0; i < 8; ++i) { mrow[u][i] = -1e30f; lrow[u][i] = 0.0f; }
  }

  const int kend = q0 + 32;  // causal: keys beyond each row masked to -1e9
  for (int k0 = 0; k0 < kend; k0 += 32) {
    // ---- K fragments for two 16-key tiles (shared by both query tiles) ----
    bf16x16 bk[2][2];
#pragma unroll
    for (int t = 0; t < 2; ++t) {
      const bf16* krow = Kp + (size_t)(k0 + t * 16 + row) * KDK;
      bk[t][0] = join8(*(const bf16x8*)(krow + half * 8),
                       *(const bf16x8*)(krow + 16 + half * 8));
      bk[t][1] = join8(*(const bf16x8*)(krow + 32 + half * 8),
                       *(const bf16x8*)(krow + 48 + half * 8));
    }

    f32x8 sc[2][2];  // [query tile][key tile]
#pragma unroll
    for (int u = 0; u < 2; ++u)
#pragma unroll
      for (int t = 0; t < 2; ++t) {
        f32x8 s = {0.f, 0.f, 0.f, 0.f, 0.f, 0.f, 0.f, 0.f};
        s = wmma_bf16(aq[u][0], bk[t][0], s);
        s = wmma_bf16(aq[u][1], bk[t][1], s);
        // scale + causal mask (masked entries get -1e9, like the reference)
#pragma unroll
        for (int i = 0; i < 8; ++i) {
          int qg = q0 + u * 16 + i + 8 * half;
          int kg = k0 + t * 16 + row;
          float val = s[i] * ATT_SCALE;
          s[i] = (kg > qg) ? -1e9f : val;
        }
        sc[u][t] = s;
      }

    // ---- online softmax (row reductions across the 16-lane group) ----
#pragma unroll
    for (int u = 0; u < 2; ++u) {
      float alpha[8];
#pragma unroll
      for (int i = 0; i < 8; ++i) {
        float r = fmaxf(sc[u][0][i], sc[u][1][i]);
#pragma unroll
        for (int mo = 1; mo < 16; mo <<= 1) r = fmaxf(r, __shfl_xor(r, mo, 32));
        float mn = fmaxf(mrow[u][i], r);
        alpha[i] = __expf(mrow[u][i] - mn);
        mrow[u][i] = mn;
      }
#pragma unroll
      for (int t = 0; t < 2; ++t)
#pragma unroll
        for (int i = 0; i < 8; ++i)
          sc[u][t][i] = __expf(sc[u][t][i] - mrow[u][i]);
#pragma unroll
      for (int i = 0; i < 8; ++i) {
        float r = sc[u][0][i] + sc[u][1][i];
#pragma unroll
        for (int mo = 1; mo < 16; mo <<= 1) r += __shfl_xor(r, mo, 32);
        lrow[u][i] = lrow[u][i] * alpha[i] + r;
      }
#pragma unroll
      for (int nt = 0; nt < 4; ++nt)
#pragma unroll
        for (int i = 0; i < 8; ++i) acc[u][nt][i] *= alpha[i];
    }

    // ---- relayout P (C layout -> A layout) through wave-private LDS ----
#pragma unroll
    for (int u = 0; u < 2; ++u) {
      bf16* pp = &ldsP[wave][u * 512];
#pragma unroll
      for (int t = 0; t < 2; ++t)
#pragma unroll
        for (int i = 0; i < 8; ++i)
          pp[(i + 8 * half) * 32 + t * 16 + row] = (bf16)sc[u][t][i];
    }
    asm volatile("s_wait_dscnt 0" ::: "memory");
    bf16x16 ap[2];
#pragma unroll
    for (int u = 0; u < 2; ++u) {
      const bf16* prow = &ldsP[wave][u * 512] + row * 32;
      ap[u] = join8(*(const bf16x8*)(prow + half * 8),
                    *(const bf16x8*)(prow + 16 + half * 8));
    }

    // ---- acc += P(16x32) @ V(32x64): V fragment reused by both q tiles ----
#pragma unroll
    for (int nt = 0; nt < 4; ++nt) {
      const bf16* vrow = Vp + (size_t)(nt * 16 + row) * KS + k0 + half * 8;
      bf16x16 bv = join8(*(const bf16x8*)vrow, *(const bf16x8*)(vrow + 16));
      acc[0][nt] = wmma_bf16(ap[0], bv, acc[0][nt]);
      acc[1][nt] = wmma_bf16(ap[1], bv, acc[1][nt]);
    }
  }

  // ---- normalize and write ctx in [B,S,D] bf16 for the output GEMM ----
#pragma unroll
  for (int u = 0; u < 2; ++u)
#pragma unroll
    for (int nt = 0; nt < 4; ++nt)
#pragma unroll
      for (int i = 0; i < 8; ++i) {
        int qg = q0 + u * 16 + i + 8 * half;
        float o = acc[u][nt][i] / lrow[u][i];
        Ctx[((size_t)b * KS + qg) * KD + h * 64 + nt * 16 + row] = (bf16)o;
      }
}

// ---------------------------------------------------------------------------
// Host-side orchestration (all on `stream`, workspace reused stream-ordered).
// ---------------------------------------------------------------------------
extern "C" void kernel_launch(void* const* d_in, const int* /*in_sizes*/,
                              int /*n_in*/, void* d_out, int /*out_size*/,
                              void* d_ws, size_t /*ws_size*/,
                              hipStream_t stream) {
  const float* q   = (const float*)d_in[0];
  const float* k   = (const float*)d_in[1];
  const float* v   = (const float*)d_in[2];
  // d_in[3] = mask: constant causal tril, implemented analytically
  const float* w_q = (const float*)d_in[4];
  const float* b_q = (const float*)d_in[5];
  const float* w_k = (const float*)d_in[6];
  const float* b_k = (const float*)d_in[7];
  const float* w_v = (const float*)d_in[8];
  const float* b_v = (const float*)d_in[9];
  const float* w_o = (const float*)d_in[10];
  const float* b_o = (const float*)d_in[11];

  const size_t MD = (size_t)KM * KD;   // 8M elems
  const size_t DDn = (size_t)KD * KD;  // 1M elems

  char* p = (char*)d_ws;
  auto take = [&](size_t bytes) {
    char* r = p;
    p += (bytes + 255) & ~(size_t)255;
    return r;
  };
  bf16* Xb  = (bf16*)take(MD * 2);   // bf16 staging for X / ctx (reused)
  bf16* Wq  = (bf16*)take(DDn * 2);
  bf16* Wk  = (bf16*)take(DDn * 2);
  bf16* Wv  = (bf16*)take(DDn * 2);
  bf16* Wo  = (bf16*)take(DDn * 2);
  float* Pf = (float*)take(MD * 4);  // fp32 projection output (reused 3x)
  bf16* Qbuf = (bf16*)take(MD * 2);  // [B,H,S,64]
  bf16* Kbuf = (bf16*)take(MD * 2);  // [B,H,S,64]
  bf16* Vtb  = (bf16*)take(MD * 2);  // [B,H,64,S]

  const dim3 blk(256);
  // weights -> bf16
  {
    int n4 = (int)(DDn / 4);
    dim3 g((n4 + 255) / 256);
    cvt_f32_bf16_x4<<<g, blk, 0, stream>>>((const float4*)w_q, (bf16x4*)Wq, n4);
    cvt_f32_bf16_x4<<<g, blk, 0, stream>>>((const float4*)w_k, (bf16x4*)Wk, n4);
    cvt_f32_bf16_x4<<<g, blk, 0, stream>>>((const float4*)w_v, (bf16x4*)Wv, n4);
    cvt_f32_bf16_x4<<<g, blk, 0, stream>>>((const float4*)w_o, (bf16x4*)Wo, n4);
  }

  const dim3 gemm_grid(KM / 32 / 8, KD / 64);       // 32 x 16
  const dim3 rope_grid((unsigned)(MD / 256));       // 32768
  const int xn4 = (int)(MD / 4);
  const dim3 cvt_grid((xn4 + 255) / 256);

  // Q projection + RoPE
  cvt_f32_bf16_x4<<<cvt_grid, blk, 0, stream>>>((const float4*)q, (bf16x4*)Xb, xn4);
  gemm_bf16_nt<<<gemm_grid, blk, 0, stream>>>(Xb, Wq, b_q, Pf);
  rope_pack<<<rope_grid, blk, 0, stream>>>(Pf, Qbuf, 1);
  // K projection + RoPE
  cvt_f32_bf16_x4<<<cvt_grid, blk, 0, stream>>>((const float4*)k, (bf16x4*)Xb, xn4);
  gemm_bf16_nt<<<gemm_grid, blk, 0, stream>>>(Xb, Wk, b_k, Pf);
  rope_pack<<<rope_grid, blk, 0, stream>>>(Pf, Kbuf, 1);
  // V projection + transpose pack
  cvt_f32_bf16_x4<<<cvt_grid, blk, 0, stream>>>((const float4*)v, (bf16x4*)Xb, xn4);
  gemm_bf16_nt<<<gemm_grid, blk, 0, stream>>>(Xb, Wv, b_v, Pf);
  rope_pack<<<rope_grid, blk, 0, stream>>>(Pf, Vtb, 0);

  // attention: ctx (bf16) written into Xb (free after V projection)
  flash_attn<<<dim3(KS / 32 / 8, KB * KH), blk, 0, stream>>>(Qbuf, Kbuf, Vtb, Xb);

  // output projection -> fp32 d_out
  gemm_bf16_nt<<<gemm_grid, blk, 0, stream>>>(Xb, Wo, b_o, (float*)d_out);
}
